// MoE_4355096838544
// MI455X (gfx1250) — compile-verified, matrix operand order checked by reading, not run on hardware
//
#include <hip/hip_runtime.h>
#include <hip/hip_bf16.h>
#include <cstdint>

// ---------------------------------------------------------------------------
// MoE reference collapses to:
//   counts[e] = #times expert e appears in global top-2 of gate logits
//   out       = sum_e (counts[e]/32768) * relu(X @ W_e^T + b_e)
// X=[16384,1024] fp32, W=[8,1024,1024] fp32.  Compute bound (AI~1700 FLOP/B).
//
// Strategy: bf16x3 emulation of fp32 GEMM on v_wmma_f32_16x16x32_bf16.
//  Phase 1: split X,W into bf16 hi/lo ONCE, stored in d_ws pre-swizzled in
//           WMMA fragment order: [kchunk][16-row tile][hi|lo][lane][16 halves].
//  Phase 2: GEMM stages 16KB tiles to LDS with global_load_async_to_lds_b128
//           (double buffered, s_wait_asynccnt), fragments = 2x ds_load_b128,
//           24 WMMAs per K-chunk per wave.  Fallback fused kernel if ws small.
// ---------------------------------------------------------------------------

typedef __attribute__((ext_vector_type(16))) __bf16 v16bf;
typedef __attribute__((ext_vector_type(8)))  float  v8f;
typedef __attribute__((ext_vector_type(4)))  __bf16 bf16x4;

#define M_TOK   16384
#define DDIM    1024
#define NEXP    8
#define TOTALK  32768.0f
#define LDS_K   36          // fallback-kernel LDS stride

// workspace layout (bytes): [0,32) counts, [256, 256+64MB) Xsw, then 32MB Wsw
#define XSW_ELEMS  33554432ull   // 16384*1024*2(parts)
#define WSW_ELEMS  16777216ull   // 8*1024*1024*2(parts)
#define WS_NEED    (256ull + XSW_ELEMS * 2 + WSW_ELEMS * 2)

// ------------------------------ gating ------------------------------------

__global__ void moe_zero_counts(float* counts) {
    if (threadIdx.x < NEXP) counts[threadIdx.x] = 0.0f;
}

__global__ __launch_bounds__(256) void moe_gate_topk(
    const float* __restrict__ x, const float* __restrict__ gate_w,
    const float* __restrict__ gate_b, float* __restrict__ counts)
{
    const int lane = threadIdx.x & 31;
    const int wave = threadIdx.x >> 5;
    const int tok  = blockIdx.x * 8 + wave;

    const float4* x4 = (const float4*)(x + (size_t)tok * DDIM);
    const float4* g4 = (const float4*)gate_w;

    float acc[NEXP];
#pragma unroll
    for (int e = 0; e < NEXP; ++e) acc[e] = 0.0f;
#pragma unroll
    for (int i = 0; i < 8; ++i) {
        const int c = lane + i * 32;
        const float4 xv = x4[c];
#pragma unroll
        for (int e = 0; e < NEXP; ++e) {
            const float4 gv = g4[e * (DDIM / 4) + c];
            acc[e] += xv.x * gv.x + xv.y * gv.y + xv.z * gv.z + xv.w * gv.w;
        }
    }
    float logit[NEXP];
#pragma unroll
    for (int e = 0; e < NEXP; ++e) {
        float v = acc[e];
#pragma unroll
        for (int off = 16; off > 0; off >>= 1) v += __shfl_xor(v, off, 32);
        logit[e] = v + gate_b[e];
    }
    if (lane == 0) {
        int b1 = 0; float v1 = logit[0];
#pragma unroll
        for (int e = 1; e < NEXP; ++e)
            if (logit[e] > v1) { v1 = logit[e]; b1 = e; }
        int b2 = -1; float v2 = -3.4e38f;
#pragma unroll
        for (int e = 0; e < NEXP; ++e) {
            if (e == b1) continue;
            if (logit[e] > v2) { v2 = logit[e]; b2 = e; }
        }
        atomicAdd(&counts[b1], 1.0f);
        atomicAdd(&counts[b2], 1.0f);
    }
}

// --------------------- phase 1: bf16 hi/lo pre-swizzle ---------------------
// Fragment layout per 16-row tile & 32-K chunk (ISA 7.12.2, 16-bit operands):
//   lane = (row&15) + 16*((k>>3)&1);  half = (k&7) + 8*(k>>4)
// Unit (1KB) = [lane 0..31][16 bf16 halves]; a float4 (4 consecutive k) maps
// to 4 consecutive halves of one lane -> one b64 store per part.

__device__ __forceinline__ void split4(float4 f, bf16x4& hi, bf16x4& lo) {
    __bf16 h0 = (__bf16)f.x, h1 = (__bf16)f.y, h2 = (__bf16)f.z, h3 = (__bf16)f.w;
    hi = (bf16x4){h0, h1, h2, h3};
    lo = (bf16x4){(__bf16)(f.x - (float)h0), (__bf16)(f.y - (float)h1),
                  (__bf16)(f.z - (float)h2), (__bf16)(f.w - (float)h3)};
}

__global__ __launch_bounds__(256) void moe_convert_x(
    const float* __restrict__ x, __bf16* __restrict__ xsw)
{
    const int gid = blockIdx.x * 256 + threadIdx.x;    // 4M float4 total
    const int m  = gid >> 8;             // token row
    const int k  = (gid & 255) << 2;     // column (multiple of 4)
    const int kk = k & 31, cc = k >> 5;
    const int tm = m >> 4, r = m & 15;
    const int lane = r + (((kk >> 3) & 1) << 4);
    const int h0   = (kk & 7) + ((kk >> 4) << 3);

    float4 f = *(const float4*)(x + (size_t)m * DDIM + k);
    bf16x4 hi, lo;
    split4(f, hi, lo);
    const size_t base = ((size_t)(cc * 1024 + tm) * 2) * 512 + lane * 16 + h0;
    *(bf16x4*)(xsw + base)       = hi;   // part 0
    *(bf16x4*)(xsw + base + 512) = lo;   // part 1
}

__global__ __launch_bounds__(256) void moe_convert_w(
    const float* __restrict__ w, __bf16* __restrict__ wsw)
{
    const int gid = blockIdx.x * 256 + threadIdx.x;    // 2M float4 total
    const int row = gid >> 8;            // 0..8191 : e*1024 + n
    const int e   = row >> 10;
    const int n   = row & 1023;
    const int k   = (gid & 255) << 2;
    const int kk = k & 31, cc = k >> 5;
    const int tn = n >> 4, r = n & 15;
    const int lane = r + (((kk >> 3) & 1) << 4);
    const int h0   = (kk & 7) + ((kk >> 4) << 3);

    float4 f = *(const float4*)(w + (size_t)row * DDIM + k);
    bf16x4 hi, lo;
    split4(f, hi, lo);
    const size_t base = (size_t)e * 2097152 +
                        ((size_t)(cc * 64 + tn) * 2) * 512 + lane * 16 + h0;
    *(bf16x4*)(wsw + base)       = hi;
    *(bf16x4*)(wsw + base + 512) = lo;
}

// --------------------- phase 2: async-staged WMMA GEMM ---------------------

__device__ __forceinline__ v16bf load_frag(const char* buf, int t, int part,
                                           int lane) {
    const char* p = buf + (t * 2 + part) * 1024 + lane * 32;
    union { v16bf f; uint4 q[2]; } u;
    u.q[0] = *(const uint4*)p;
    u.q[1] = *(const uint4*)(p + 16);
    return u.f;
}

__global__ __launch_bounds__(256, 1) void moe_expert_gemm_sw(
    const __bf16* __restrict__ xsw, const __bf16* __restrict__ wsw,
    const float* __restrict__ expert_b, const float* __restrict__ counts,
    float* __restrict__ out)
{
    __shared__ __align__(16) char XsB[2][16384];
    __shared__ __align__(16) char WsB[2][16384];

    const int tid  = threadIdx.x;
    const int lane = tid & 31;
    const int wave = tid >> 5;
    const int wm   = wave & 3;            // 4 waves down M (2 tiles each)
    const int wn   = wave >> 2;           // 2 waves across N (4 tiles each)
    const int mb8  = blockIdx.x * 8;      // first 16-row tile of this block
    const int nb8  = blockIdx.y * 8;

    // issue one chunk's async copies: 16KB X + 16KB W, 4+4 b128 per thread
    auto issue = [&](int it, int buf) {
        const int e = it >> 5, c = it & 31;
        const char* xsrc = (const char*)xsw + ((size_t)(c * 1024 + mb8)) * 2048;
        const char* wsrc = (const char*)wsw + ((size_t)e << 22) +
                           ((size_t)(c * 64 + nb8)) * 2048;
#pragma unroll
        for (int j = 0; j < 4; ++j) {
            const int off = tid * 16 + j * 4096;
            asm volatile("global_load_async_to_lds_b128 %0, %1, off"
                         :: "v"((unsigned)(uintptr_t)(XsB[buf] + off)),
                            "v"((unsigned long long)(uintptr_t)(xsrc + off))
                         : "memory");
            asm volatile("global_load_async_to_lds_b128 %0, %1, off"
                         :: "v"((unsigned)(uintptr_t)(WsB[buf] + off)),
                            "v"((unsigned long long)(uintptr_t)(wsrc + off))
                         : "memory");
        }
    };

    v8f fin[2][4], acc[2][4];
#pragma unroll
    for (int mt = 0; mt < 2; ++mt)
#pragma unroll
        for (int nt = 0; nt < 4; ++nt) {
            fin[mt][nt] = (v8f)0.0f;
            acc[mt][nt] = (v8f)0.0f;
        }

    issue(0, 0);

    for (int it = 0; it < NEXP * 32; ++it) {
        const int cb = it & 1;
        const bool more = (it + 1) < NEXP * 32;
        if (more) {
            issue(it + 1, (it + 1) & 1);
            asm volatile("s_wait_asynccnt 8" ::: "memory");   // cur chunk done
        } else {
            asm volatile("s_wait_asynccnt 0" ::: "memory");
        }
        __syncthreads();

        v16bf ahi[2], alo[2], bhi[4], blo[4];
#pragma unroll
        for (int mt = 0; mt < 2; ++mt) {
            ahi[mt] = load_frag(XsB[cb], wm * 2 + mt, 0, lane);
            alo[mt] = load_frag(XsB[cb], wm * 2 + mt, 1, lane);
        }
#pragma unroll
        for (int nt = 0; nt < 4; ++nt) {
            bhi[nt] = load_frag(WsB[cb], wn * 4 + nt, 0, lane);
            blo[nt] = load_frag(WsB[cb], wn * 4 + nt, 1, lane);
        }
#pragma unroll
        for (int mt = 0; mt < 2; ++mt)
#pragma unroll
            for (int nt = 0; nt < 4; ++nt) {
                acc[mt][nt] = __builtin_amdgcn_wmma_f32_16x16x32_bf16(
                    false, ahi[mt], false, bhi[nt], (short)0, acc[mt][nt],
                    false, false);
                acc[mt][nt] = __builtin_amdgcn_wmma_f32_16x16x32_bf16(
                    false, ahi[mt], false, blo[nt], (short)0, acc[mt][nt],
                    false, false);
                acc[mt][nt] = __builtin_amdgcn_wmma_f32_16x16x32_bf16(
                    false, alo[mt], false, bhi[nt], (short)0, acc[mt][nt],
                    false, false);
            }
        __syncthreads();   // protect buffer cb before it is re-filled

        if ((it & 31) == 31) {          // expert finished: relu/scale/acc
            const int e = it >> 5;
            const float scale = counts[e] * (1.0f / TOTALK);
#pragma unroll
            for (int nt = 0; nt < 4; ++nt) {
                const int ncol = nb8 * 16 + wn * 64 + nt * 16 + (lane & 15);
                const float bias = expert_b[e * DDIM + ncol];
#pragma unroll
                for (int mt = 0; mt < 2; ++mt)
#pragma unroll
                    for (int r = 0; r < 8; ++r) {
                        fin[mt][nt][r] +=
                            scale * fmaxf(acc[mt][nt][r] + bias, 0.0f);
                        if (r == 7) acc[mt][nt] = (v8f)0.0f;
                    }
            }
        }
    }

#pragma unroll
    for (int mt = 0; mt < 2; ++mt)
#pragma unroll
        for (int nt = 0; nt < 4; ++nt) {
            const int ncol = nb8 * 16 + wn * 64 + nt * 16 + (lane & 15);
#pragma unroll
            for (int r = 0; r < 8; ++r) {
                const int mrow =
                    mb8 * 16 + wm * 32 + mt * 16 + ((lane >> 4) << 3) + r;
                out[(size_t)mrow * DDIM + ncol] = fin[mt][nt][r];
            }
        }
}

// -------------------- fallback fused kernel (small ws) ---------------------

__device__ __forceinline__ void make_frag(const float* __restrict__ tile,
                                          int lane, v16bf& hi, v16bf& lo) {
    const int r = lane & 15;
    const int grp = (lane >> 4) << 3;
    const float* row = tile + r * LDS_K;
#pragma unroll
    for (int p = 0; p < 8; ++p) {
        const int k0 = 2 * p + ((p >= 4) ? 8 : 0) + grp;
        const float2 v = *(const float2*)(row + k0);
        const __bf16 h0 = (__bf16)v.x;
        const __bf16 h1 = (__bf16)v.y;
        hi[2 * p] = h0; hi[2 * p + 1] = h1;
        lo[2 * p] = (__bf16)(v.x - (float)h0);
        lo[2 * p + 1] = (__bf16)(v.y - (float)h1);
    }
}

__global__ __launch_bounds__(256, 1) void moe_expert_gemm_fused(
    const float* __restrict__ x, const float* __restrict__ expert_w,
    const float* __restrict__ expert_b, const float* __restrict__ counts,
    float* __restrict__ out)
{
    __shared__ float Xs[128 * LDS_K];
    __shared__ float Ws[128 * LDS_K];

    const int tid = threadIdx.x, lane = tid & 31, wave = tid >> 5;
    const int wm = wave & 3, wn = wave >> 2;
    const int m0 = blockIdx.x * 128, n0 = blockIdx.y * 128;

    v8f fin[2][4];
#pragma unroll
    for (int mt = 0; mt < 2; ++mt)
#pragma unroll
        for (int nt = 0; nt < 4; ++nt) fin[mt][nt] = (v8f)0.0f;

    for (int e = 0; e < NEXP; ++e) {
        const float* wbase = expert_w + ((size_t)e << 20);
        v8f acc[2][4];
#pragma unroll
        for (int mt = 0; mt < 2; ++mt)
#pragma unroll
            for (int nt = 0; nt < 4; ++nt) acc[mt][nt] = (v8f)0.0f;

        for (int k0 = 0; k0 < DDIM; k0 += 32) {
#pragma unroll
            for (int i = 0; i < 4; ++i) {
                const int idx = tid + i * 256;
                const int r = idx >> 3, c = (idx & 7) << 2;
                *(float4*)&Xs[r * LDS_K + c] =
                    *(const float4*)(x + (size_t)(m0 + r) * DDIM + k0 + c);
                *(float4*)&Ws[r * LDS_K + c] =
                    *(const float4*)(wbase + (size_t)(n0 + r) * DDIM + k0 + c);
            }
            __syncthreads();

            v16bf ahi[2], alo[2], bhi[4], blo[4];
#pragma unroll
            for (int mt = 0; mt < 2; ++mt)
                make_frag(&Xs[(wm * 32 + mt * 16) * LDS_K], lane, ahi[mt], alo[mt]);
#pragma unroll
            for (int nt = 0; nt < 4; ++nt)
                make_frag(&Ws[(wn * 64 + nt * 16) * LDS_K], lane, bhi[nt], blo[nt]);
#pragma unroll
            for (int mt = 0; mt < 2; ++mt)
#pragma unroll
                for (int nt = 0; nt < 4; ++nt) {
                    acc[mt][nt] = __builtin_amdgcn_wmma_f32_16x16x32_bf16(
                        false, ahi[mt], false, bhi[nt], (short)0, acc[mt][nt],
                        false, false);
                    acc[mt][nt] = __builtin_amdgcn_wmma_f32_16x16x32_bf16(
                        false, ahi[mt], false, blo[nt], (short)0, acc[mt][nt],
                        false, false);
                    acc[mt][nt] = __builtin_amdgcn_wmma_f32_16x16x32_bf16(
                        false, alo[mt], false, bhi[nt], (short)0, acc[mt][nt],
                        false, false);
                }
            __syncthreads();
        }

        const float scale = counts[e] * (1.0f / TOTALK);
#pragma unroll
        for (int nt = 0; nt < 4; ++nt) {
            const int ncol = n0 + wn * 64 + nt * 16 + (lane & 15);
            const float bias = expert_b[e * DDIM + ncol];
#pragma unroll
            for (int mt = 0; mt < 2; ++mt)
#pragma unroll
                for (int r = 0; r < 8; ++r)
                    fin[mt][nt][r] += scale * fmaxf(acc[mt][nt][r] + bias, 0.0f);
        }
    }
#pragma unroll
    for (int mt = 0; mt < 2; ++mt)
#pragma unroll
        for (int nt = 0; nt < 4; ++nt) {
            const int ncol = n0 + wn * 64 + nt * 16 + (lane & 15);
#pragma unroll
            for (int r = 0; r < 8; ++r) {
                const int mrow = m0 + wm * 32 + mt * 16 + ((lane >> 4) << 3) + r;
                out[(size_t)mrow * DDIM + ncol] = fin[mt][nt][r];
            }
        }
}

// ------------------------------ launch ------------------------------------

extern "C" void kernel_launch(void* const* d_in, const int* in_sizes, int n_in,
                              void* d_out, int out_size, void* d_ws, size_t ws_size,
                              hipStream_t stream) {
    const float* x        = (const float*)d_in[0];
    const float* gate_w   = (const float*)d_in[1];
    const float* gate_b   = (const float*)d_in[2];
    const float* expert_w = (const float*)d_in[3];
    const float* expert_b = (const float*)d_in[4];
    float*       out      = (float*)d_out;
    float*       counts   = (float*)d_ws;

    moe_zero_counts<<<1, 32, 0, stream>>>(counts);
    moe_gate_topk<<<M_TOK / 8, 256, 0, stream>>>(x, gate_w, gate_b, counts);

    dim3 grid(M_TOK / 128, DDIM / 128);
    if (ws_size >= WS_NEED) {
        __bf16* xsw = (__bf16*)((char*)d_ws + 256);
        __bf16* wsw = xsw + XSW_ELEMS;
        moe_convert_x<<<(M_TOK * DDIM / 4) / 256, 256, 0, stream>>>(x, xsw);
        moe_convert_w<<<(NEXP * DDIM * DDIM / 4) / 256, 256, 0, stream>>>(expert_w, wsw);
        moe_expert_gemm_sw<<<grid, 256, 0, stream>>>(xsw, wsw, expert_b, counts, out);
    } else {
        moe_expert_gemm_fused<<<grid, 256, 0, stream>>>(x, expert_w, expert_b,
                                                        counts, out);
    }
}
